// SchNet_model_1271310320362
// MI455X (gfx1250) — compile-verified, matrix operand order checked by reading, not run on hardware
//
#include <hip/hip_runtime.h>
#include <hip/hip_bf16.h>

// ---------------------------------------------------------------------------
// SchNet fused CDNA5 implementation (round 2).
// - Structured graph: edges [i*32, i*32+32) all have dst=i -> contiguous
//   per-node reduction, no atomics.
// - All matmuls on v_wmma_f32_16x16x32_bf16 (f32 accumulate).
// - Weights pre-swizzled into fragment-major bf16 so every B fragment is
//   2 x b128 loads; A fragments / transposes use contiguous 16B accesses.
// ---------------------------------------------------------------------------

#define NODES  32768
#define NMOLS  512
#define KNBR   32
#define EDGES  (NODES * KNBR)
#define H      128
#define F      128
#define G      50
#define GP     64          // G padded to 64 for the WMMA K-dim
#define LAYERS 6
#define WPL    (GP*F + 4*F*F)   // bf16 weight elements per layer = 73728

typedef __attribute__((ext_vector_type(16))) __bf16 v16bf;
typedef __attribute__((ext_vector_type(8)))  float  v8f;

union Frag {
  v16bf v;
  uint4 q[2];
};

__device__ __forceinline__ v8f wmma_bf16(v16bf a, v16bf b, v8f c) {
  // D = A(16x32) * B(32x16) + C(16x16), f32 accumulate
  return __builtin_amdgcn_wmma_f32_16x16x32_bf16(false, a, false, b, (short)0, c,
                                                 false, false);
}

// bf16 A/B fragment K-index for element j, given lane's K-half (kh = lane>>4).
// ISA 7.12.2: lanes 0-15 hold K=0..7 (VGPR0-3) and K=16..23 (VGPR4-7);
// lanes 16-31 hold K=8..15 and K=24..31.
__device__ __forceinline__ int fkj(int j, int kh) {
  return kh * 8 + (j < 8 ? j : j + 8);
}

// shifted softplus: softplus(x) - log(2), numerically stable
__device__ __forceinline__ float sspf(float x) {
  float ax = __builtin_fabsf(x);
  return fmaxf(x, 0.0f) + __logf(1.0f + __expf(-ax)) - 0.69314718056f;
}

// Load a pre-swizzled B fragment: 32 contiguous bytes per lane.
// Works for LDS or global pointers (2 x *_load_b128).
__device__ __forceinline__ v16bf load_bfrag(const __bf16* swz, int c, int nt,
                                            int lane) {
  const uint4* p = (const uint4*)(swz + ((((c << 3) + nt) << 5) + lane) * 16);
  Frag f;
  f.q[0] = p[0];
  f.q[1] = p[1];
  return f.v;
}

// Load an A fragment from a row-major bf16 row (length KD*32, 16B aligned):
// elements [c*32+kh*8 .. +7] and [c*32+16+kh*8 .. +7].
__device__ __forceinline__ v16bf load_afrag_row(const __bf16* row, int c,
                                                int kh) {
  const uint4* p = (const uint4*)(row + c * 32 + kh * 8);
  Frag f;
  f.q[0] = p[0];
  f.q[1] = p[2];
  return f.v;
}

// Load an A fragment from a row-major f32 row, converting to bf16.
__device__ __forceinline__ v16bf load_afrag_f32(const float* row, int c,
                                                int kh) {
  const float4* p = (const float4*)(row + c * 32 + kh * 8);
  float4 f0 = p[0], f1 = p[1], f2 = p[4], f3 = p[5];
  float t[16] = {f0.x, f0.y, f0.z, f0.w, f1.x, f1.y, f1.z, f1.w,
                 f2.x, f2.y, f2.z, f2.w, f3.x, f3.y, f3.z, f3.w};
  v16bf r;
#pragma unroll
  for (int j = 0; j < 16; ++j) r[j] = (__bf16)t[j];
  return r;
}

// ---------------------------------------------------------------------------
// Weight conversion + fragment swizzle.  Per layer, bf16 sections:
//   [w1 (KD=2) | w2 (KD=4) | conv_w1 | conv_w2 | lin_w]
// Element s of a section: frag=s/16, j=s%16; lane=frag%32; nt=(frag/32)%8;
// c=frag/256; source = W[(c*32+fkj(j,lane>>4)) * 128 + nt*16 + (lane&15)].
// ---------------------------------------------------------------------------
__global__ __launch_bounds__(256) void cvt_weights_k(
    const float* __restrict__ mw1, const float* __restrict__ mw2,
    const float* __restrict__ cw1, const float* __restrict__ cw2,
    const float* __restrict__ lw, __bf16* __restrict__ out) {
  int idx = blockIdx.x * 256 + threadIdx.x;
  if (idx >= LAYERS * WPL) return;
  int l = idx / WPL, r = idx % WPL;
  const float* W;
  int krows, s;
  if (r < GP * F) {
    W = mw1 + l * G * F;  s = r;                    krows = G;   // zero-pad 50..63
  } else if (r < GP * F + F * F) {
    W = mw2 + l * F * F;  s = r - GP * F;           krows = F;
  } else if (r < GP * F + 2 * F * F) {
    W = cw1 + l * F * F;  s = r - GP * F - F * F;   krows = F;
  } else if (r < GP * F + 3 * F * F) {
    W = cw2 + l * F * F;  s = r - GP * F - 2 * F * F; krows = F;
  } else {
    W = lw + l * F * F;   s = r - GP * F - 3 * F * F; krows = F;
  }
  int j = s & 15, frag = s >> 4;
  int lane = frag & 31, t2 = frag >> 5;
  int nt = t2 & 7, c = t2 >> 3;
  int k = c * 32 + fkj(j, lane >> 4);
  int col = nt * 16 + (lane & 15);
  float v = (k < krows) ? W[k * F + col] : 0.0f;
  out[idx] = (__bf16)v;
}

// h[i,:] = emb[z[i],:]
__global__ __launch_bounds__(256) void embed_k(
    const int* __restrict__ z, const float* __restrict__ emb,
    float* __restrict__ h) {
  int idx = blockIdx.x * 256 + threadIdx.x;   // N*H threads
  int i = idx >> 7, c = idx & 127;
  h[idx] = emb[z[i] * H + c];
}

// per-edge distance + cosine cutoff
__global__ __launch_bounds__(256) void edge_geom_k(
    const int* __restrict__ ei, const float* __restrict__ pos,
    float* __restrict__ de, float* __restrict__ ce) {
  int e = blockIdx.x * 256 + threadIdx.x;
  if (e >= EDGES) return;
  int s = ei[e], t = ei[EDGES + e];
  float dx = pos[t * 3 + 0] - pos[s * 3 + 0];
  float dy = pos[t * 3 + 1] - pos[s * 3 + 1];
  float dz = pos[t * 3 + 2] - pos[s * 3 + 2];
  float d = __builtin_sqrtf(dx * dx + dy * dy + dz * dz);
  de[e] = d;
  ce[e] = 0.5f * (__cosf(d * 3.14159265358979f * 0.1f) + 1.0f);
}

// ---------------------------------------------------------------------------
// hx = h @ conv_w1 (f32 in, bf16 out). One wave per 16-node tile.
// Output goes through the LDS tile so global stores are coalesced b128.
// ---------------------------------------------------------------------------
__global__ __launch_bounds__(128) void node_linear_k(
    const float* __restrict__ h, const __bf16* __restrict__ w,
    __bf16* __restrict__ hx) {
  __shared__ __align__(16) __bf16 wsm[H * F];      // 32 KB swizzled weights
  __shared__ __align__(16) __bf16 tls[4][16 * F];  // 16 KB per-wave tiles
  {
    const uint4* s = (const uint4*)w;
    uint4* d = (uint4*)wsm;
    for (int i = threadIdx.x; i < H * F / 8; i += 128) d[i] = s[i];
  }
  __syncthreads();
  int wave = threadIdx.x >> 5, lane = threadIdx.x & 31;
  int m = lane & 15, kh = lane >> 4, n = m;
  int tile = blockIdx.x * 4 + wave;
  const float* arow = h + (size_t)(tile * 16 + m) * H;
  v16bf A[4];
#pragma unroll
  for (int c = 0; c < 4; ++c) A[c] = load_afrag_f32(arow, c, kh);
#pragma unroll
  for (int nt = 0; nt < 8; ++nt) {
    v8f acc = {0.f, 0.f, 0.f, 0.f, 0.f, 0.f, 0.f, 0.f};
#pragma unroll
    for (int c = 0; c < 4; ++c)
      acc = wmma_bf16(A[c], load_bfrag(wsm, c, nt, lane), acc);
#pragma unroll
    for (int r = 0; r < 8; ++r)
      tls[wave][(r + 8 * kh) * F + nt * 16 + n] = (__bf16)acc[r];
  }
  // cooperative coalesced write-out: lane -> row m, half kh (128B)
  {
    const uint4* sp = (const uint4*)(&tls[wave][m * F + kh * 64]);
    uint4* dp = (uint4*)(hx + (size_t)(tile * 16 + m) * F + kh * 64);
#pragma unroll
    for (int q = 0; q < 8; ++q) dp[q] = sp[q];
  }
}

// ---------------------------------------------------------------------------
// Fused edge pipeline + aggregation. One wave owns one dst node (32 edges,
// two 16-edge groups). Per group:
//   gauss(d) -> WMMA MLP1(K=64) -> +b1 -> ssp -> LDS transpose ->
//   WMMA MLP2(K=128) -> +b2 -> *C ; hx[src] rows coop-staged into the same
//   LDS tile (DS ops are in-order per wave) -> modulate -> row reduce.
// agg[node,:] written once, no atomics.
// ---------------------------------------------------------------------------
__global__ __launch_bounds__(128) void edge_layer_k(
    const float* __restrict__ de, const float* __restrict__ ce,
    const int* __restrict__ esrc, const __bf16* __restrict__ hx,
    const __bf16* __restrict__ w1, const __bf16* __restrict__ w2,
    const float* __restrict__ b1, const float* __restrict__ b2,
    float* __restrict__ agg) {
  __shared__ __align__(16) __bf16 w1s[GP * F];     // 16 KB swizzled
  __shared__ __align__(16) __bf16 w2s[F * F];      // 32 KB swizzled
  __shared__ __align__(16) __bf16 tls[4][16 * F];  // 16 KB per-wave tiles
  {
    const uint4* s1 = (const uint4*)w1;
    uint4* d1 = (uint4*)w1s;
    for (int i = threadIdx.x; i < GP * F / 8; i += 128) d1[i] = s1[i];
    const uint4* s2 = (const uint4*)w2;
    uint4* d2 = (uint4*)w2s;
    for (int i = threadIdx.x; i < F * F / 8; i += 128) d2[i] = s2[i];
  }
  __syncthreads();
  int wave = threadIdx.x >> 5, lane = threadIdx.x & 31;
  int m = lane & 15, kh = lane >> 4, n = m;
  const float GS = 10.0f / 49.0f;       // gaussian spacing
  const float GC = -0.5f / (GS * GS);   // gaussian coeff
  for (int it = 0; it < 8; ++it) {
    int node = blockIdx.x * 32 + it * 4 + wave;
    __builtin_prefetch(&de[(size_t)node * KNBR + KNBR * 4], 0, 1);
    float aggv[8] = {0, 0, 0, 0, 0, 0, 0, 0};
    for (int g = 0; g < 2; ++g) {
      int eb = node * KNBR + g * 16;
      float dv = de[eb + m];
      int sl = esrc[eb + m];            // src row for coop staging (row m)
      float cv[8];
#pragma unroll
      for (int r = 0; r < 8; ++r) cv[r] = ce[eb + r + 8 * kh];
      // A1: 16 edges x 64 gaussian features (k >= 50 zero)
      v16bf A1[2];
#pragma unroll
      for (int c = 0; c < 2; ++c)
#pragma unroll
        for (int j = 0; j < 16; ++j) {
          int k = c * 32 + fkj(j, kh);
          float t = dv - (float)k * GS;
          float gg = (k < G) ? __expf(GC * t * t) : 0.0f;
          A1[c][j] = (__bf16)gg;
        }
      // MLP1 -> ssp -> transpose tile
#pragma unroll
      for (int nt = 0; nt < 8; ++nt) {
        v8f acc = {0.f, 0.f, 0.f, 0.f, 0.f, 0.f, 0.f, 0.f};
#pragma unroll
        for (int c = 0; c < 2; ++c)
          acc = wmma_bf16(A1[c], load_bfrag(w1s, c, nt, lane), acc);
        float bb = b1[nt * 16 + n];
#pragma unroll
        for (int r = 0; r < 8; ++r)
          tls[wave][(r + 8 * kh) * F + nt * 16 + n] = (__bf16)sspf(acc[r] + bb);
      }
      // A2 fragments (row-major tile -> A layout, 2 x b128 each)
      v16bf A2[4];
#pragma unroll
      for (int c = 0; c < 4; ++c)
        A2[c] = load_afrag_row(&tls[wave][m * F], c, kh);
      // coop-stage the 16 hx[src] rows into the (now dead) tile, coalesced.
      // DS ops from one wave are in-order, so A2 reads above see old data.
      {
        const uint4* sp = (const uint4*)(hx + (size_t)sl * F + kh * 64);
        uint4* dp = (uint4*)(&tls[wave][m * F + kh * 64]);
#pragma unroll
        for (int q = 0; q < 8; ++q) dp[q] = sp[q];
      }
      // MLP2 -> +b2 -> *C -> modulate by staged hx -> reduce over rows
#pragma unroll
      for (int nt = 0; nt < 8; ++nt) {
        v8f acc = {0.f, 0.f, 0.f, 0.f, 0.f, 0.f, 0.f, 0.f};
#pragma unroll
        for (int c = 0; c < 4; ++c)
          acc = wmma_bf16(A2[c], load_bfrag(w2s, c, nt, lane), acc);
        float bb = b2[nt * 16 + n];
        float part = 0.0f;
#pragma unroll
        for (int r = 0; r < 8; ++r) {
          float hv = (float)tls[wave][(r + 8 * kh) * F + nt * 16 + n];
          part += (acc[r] + bb) * cv[r] * hv;
        }
        part += __shfl_xor(part, 16, 32);   // combine the two M halves
        aggv[nt] += part;
      }
    }
    if (lane < 16) {
#pragma unroll
      for (int nt = 0; nt < 8; ++nt)
        agg[(size_t)node * F + nt * 16 + lane] = aggv[nt];
    }
  }
}

// ---------------------------------------------------------------------------
// v = ssp(agg @ conv_w2 + b) @ lin_w + lb ; h += v.  One wave per 16 nodes.
// conv_w2 staged in LDS; lin_w (swizzled bf16) streamed from L1/L2 as b128.
// Residual applied through an f32 LDS tile so the h RMW is float4-coalesced.
// ---------------------------------------------------------------------------
__global__ __launch_bounds__(128) void node_update_k(
    const float* __restrict__ agg, const __bf16* __restrict__ wA,
    const __bf16* __restrict__ wB, const float* __restrict__ bA,
    const float* __restrict__ bB, float* __restrict__ h) {
  __shared__ __align__(16) __bf16 wsm[F * H];      // 32 KB swizzled conv_w2
  __shared__ __align__(16) float  tlf[4][16 * H];  // 32 KB per-wave f32 tiles
  {
    const uint4* s = (const uint4*)wA;
    uint4* d = (uint4*)wsm;
    for (int i = threadIdx.x; i < F * H / 8; i += 128) d[i] = s[i];
  }
  __syncthreads();
  int wave = threadIdx.x >> 5, lane = threadIdx.x & 31;
  int m = lane & 15, kh = lane >> 4, n = m;
  int tile = blockIdx.x * 4 + wave;
  __bf16* tlb = (__bf16*)&tlf[wave][0];   // bf16 view (first half) for stage 1
  const float* arow = agg + (size_t)(tile * 16 + m) * F;
  v16bf A[4];
#pragma unroll
  for (int c = 0; c < 4; ++c) A[c] = load_afrag_f32(arow, c, kh);
#pragma unroll
  for (int nt = 0; nt < 8; ++nt) {
    v8f acc = {0.f, 0.f, 0.f, 0.f, 0.f, 0.f, 0.f, 0.f};
#pragma unroll
    for (int c = 0; c < 4; ++c)
      acc = wmma_bf16(A[c], load_bfrag(wsm, c, nt, lane), acc);
    float bb = bA[nt * 16 + n];
#pragma unroll
    for (int r = 0; r < 8; ++r)
      tlb[(r + 8 * kh) * H + nt * 16 + n] = (__bf16)sspf(acc[r] + bb);
  }
  v16bf A2[4];
#pragma unroll
  for (int c = 0; c < 4; ++c) A2[c] = load_afrag_row(tlb + m * H, c, kh);
  // second matmul; stash v as f32 into the tile (overwrites bf16 stage,
  // safe: DS in-order per wave and A2 already extracted)
#pragma unroll
  for (int nt = 0; nt < 8; ++nt) {
    v8f acc = {0.f, 0.f, 0.f, 0.f, 0.f, 0.f, 0.f, 0.f};
#pragma unroll
    for (int c = 0; c < 4; ++c)
      acc = wmma_bf16(A2[c], load_bfrag(wB, c, nt, lane), acc);
    float bb = bB[nt * 16 + n];
#pragma unroll
    for (int r = 0; r < 8; ++r)
      tlf[wave][(r + 8 * kh) * H + nt * 16 + n] = acc[r] + bb;
  }
  // coalesced residual: lane -> row m, half kh (64 floats = 16 float4)
  {
    const float4* tp = (const float4*)(&tlf[wave][m * H + kh * 64]);
    float4* hp = (float4*)(h + (size_t)(tile * 16 + m) * H + kh * 64);
#pragma unroll
    for (int q = 0; q < 16; ++q) {
      float4 tv = tp[q];
      float4 hv = hp[q];
      hv.x += tv.x; hv.y += tv.y; hv.z += tv.z; hv.w += tv.w;
      hp[q] = hv;
    }
  }
}

// ---------------------------------------------------------------------------
// Readout: o_i = ssp(h_i @ w1 + b1) @ w2 + b2 ; out[mol] += o_i.
// One wave per node (lane covers 2 of 64 hidden units). Negligible FLOPs.
// ---------------------------------------------------------------------------
__global__ __launch_bounds__(256) void readout_k(
    const float* __restrict__ h, const int* __restrict__ batch,
    const float* __restrict__ w1, const float* __restrict__ b1,
    const float* __restrict__ w2, const float* __restrict__ b2,
    float* __restrict__ out) {
  int wave = threadIdx.x >> 5, lane = threadIdx.x & 31;
  int node = blockIdx.x * 8 + wave;
  float acc = 0.0f;
#pragma unroll
  for (int cc = 0; cc < 2; ++cc) {
    int c = lane * 2 + cc;
    float s = b1[c];
    for (int k = 0; k < H; ++k) s += h[(size_t)node * H + k] * w1[k * 64 + c];
    acc += sspf(s) * w2[c];
  }
  for (int off = 16; off > 0; off >>= 1) acc += __shfl_down(acc, off, 32);
  if (lane == 0) atomicAdd(&out[batch[node]], acc + b2[0]);
}

// ---------------------------------------------------------------------------
extern "C" void kernel_launch(void* const* d_in, const int* in_sizes, int n_in,
                              void* d_out, int out_size, void* d_ws,
                              size_t ws_size, hipStream_t stream) {
  (void)in_sizes; (void)n_in; (void)out_size; (void)ws_size;
  const int*   z       = (const int*)  d_in[0];
  const float* pos     = (const float*)d_in[1];
  const int*   batch   = (const int*)  d_in[2];
  const int*   ei      = (const int*)  d_in[3];
  const float* emb     = (const float*)d_in[4];
  const float* mlp_w1  = (const float*)d_in[5];
  const float* mlp_b1  = (const float*)d_in[6];
  const float* mlp_w2  = (const float*)d_in[7];
  const float* mlp_b2  = (const float*)d_in[8];
  const float* conv_w1 = (const float*)d_in[9];
  const float* conv_w2 = (const float*)d_in[10];
  const float* conv_b2 = (const float*)d_in[11];
  const float* lin_w   = (const float*)d_in[12];
  const float* lin_b   = (const float*)d_in[13];
  const float* out_w1  = (const float*)d_in[14];
  const float* out_b1  = (const float*)d_in[15];
  const float* out_w2  = (const float*)d_in[16];
  const float* out_b2  = (const float*)d_in[17];
  float* out = (float*)d_out;

  char* ws = (char*)d_ws;
  size_t off = 0;
  float*  h   = (float*) (ws + off); off += (size_t)NODES * H * 4;   // 16 MB
  float*  agg = (float*) (ws + off); off += (size_t)NODES * F * 4;   // 16 MB
  __bf16* hx  = (__bf16*)(ws + off); off += (size_t)NODES * F * 2;   //  8 MB
  float*  de  = (float*) (ws + off); off += (size_t)EDGES * 4;       //  4 MB
  float*  ce  = (float*) (ws + off); off += (size_t)EDGES * 4;       //  4 MB
  __bf16* wbf = (__bf16*)(ws + off); off += (size_t)LAYERS * WPL * 2;

  hipMemsetAsync(d_out, 0, NMOLS * sizeof(float), stream);
  cvt_weights_k<<<(LAYERS * WPL + 255) / 256, 256, 0, stream>>>(
      mlp_w1, mlp_w2, conv_w1, conv_w2, lin_w, wbf);
  embed_k<<<NODES * H / 256, 256, 0, stream>>>(z, emb, h);
  edge_geom_k<<<EDGES / 256, 256, 0, stream>>>(ei, pos, de, ce);

  for (int l = 0; l < LAYERS; ++l) {
    const __bf16* w1b  = wbf + (size_t)l * WPL;
    const __bf16* w2b  = w1b + GP * F;
    const __bf16* cw1b = w1b + GP * F + 1 * F * F;
    const __bf16* cw2b = w1b + GP * F + 2 * F * F;
    const __bf16* lwb  = w1b + GP * F + 3 * F * F;
    node_linear_k<<<NODES / 64, 128, 0, stream>>>(h, cw1b, hx);
    edge_layer_k<<<NODES / 32, 128, 0, stream>>>(
        de, ce, ei, hx, w1b, w2b, mlp_b1 + l * F, mlp_b2 + l * F, agg);
    node_update_k<<<NODES / 64, 128, 0, stream>>>(
        agg, cw2b, lwb, conv_b2 + l * H, lin_b + l * H, h);
  }
  readout_k<<<NODES / 8, 256, 0, stream>>>(h, batch, out_w1, out_b1, out_w2,
                                           out_b2, out);
}
